// Net_83880711291174
// MI455X (gfx1250) — compile-verified, matrix operand order checked by reading, not run on hardware
//
#include <hip/hip_runtime.h>
#include <hip/hip_bf16.h>
#include <math.h>

// ---------------------------------------------------------------------------
// Problem constants (fixed by the reference)
// ---------------------------------------------------------------------------
#define KSP   5
#define KC    125
#define N0_   32768
#define N1_   8192
#define N2_   2048
#define N3_   512
#define NV_   64
#define E0_   262144
#define E1_   65536
#define E2_   16384
#define E3_   4096
#define C1_   32
#define C2_   64
#define C3_   128
#define C4_   256

typedef _Float16 f16x16 __attribute__((ext_vector_type(16)));
typedef _Float16 f16x8  __attribute__((ext_vector_type(8)));
typedef float    f32x8  __attribute__((ext_vector_type(8)));

// ---------------------------------------------------------------------------
// Small utility kernels
// ---------------------------------------------------------------------------
__global__ void fill_f32(float* __restrict__ p, float v, int n) {
    int t = blockIdx.x * blockDim.x + threadIdx.x;
    if (t < n) p[t] = v;
}
__global__ void fill_u32(unsigned* __restrict__ p, unsigned v, int n) {
    int t = blockIdx.x * blockDim.x + threadIdx.x;
    if (t < n) p[t] = v;
}
__global__ void fill_i32(int* __restrict__ p, int v, int n) {
    int t = blockIdx.x * blockDim.x + threadIdx.x;
    if (t < n) p[t] = v;
}
__global__ void fill_rec(int4* __restrict__ p, int n) {
    int t = blockIdx.x * blockDim.x + threadIdx.x;
    if (t < n) p[t] = make_int4(0, 0, 0, 0);  // src=0,dst=0,k=0,bw=0.0f
}
__global__ void cvt_f16(const float* __restrict__ x, _Float16* __restrict__ y, int n) {
    int t = blockIdx.x * blockDim.x + threadIdx.x;
    if (t < n) y[t] = (_Float16)x[t];
}
// W[k][Cin][Cout] (f32) -> Wt[k][Cout][Cin] (f16): B operand K-run contiguous
__global__ void w_cvt_t(const float* __restrict__ W, _Float16* __restrict__ Wt,
                        int Cin, int Cout, int total) {
    int t = blockIdx.x * blockDim.x + threadIdx.x;
    if (t >= total) return;
    int cc = Cin * Cout;
    int k  = t / cc;
    int r  = t - k * cc;
    int ci = r / Cout;
    int co = r - ci * Cout;
    Wt[((size_t)k * Cout + co) * Cin + ci] = (_Float16)W[t];
}

// ---------------------------------------------------------------------------
// Trilinear open-B-spline basis (degree 1): 8 corners, weights & kernel ids
// ---------------------------------------------------------------------------
__device__ __forceinline__ void edge_basis(const float* __restrict__ ps, int e,
                                           float (&bw)[8], int (&kx)[8]) {
    float fr[3]; int lo[3];
#pragma unroll
    for (int d = 0; d < 3; ++d) {
        float v  = ps[e * 3 + d] * (float)(KSP - 1);
        float fl = floorf(v);
        fl = fminf(fmaxf(fl, 0.0f), (float)(KSP - 2));
        lo[d] = (int)fl;
        fr[d] = v - fl;
    }
#pragma unroll
    for (int s = 0; s < 8; ++s) {
        float w = 1.0f; int k = 0; int mul = 1;
#pragma unroll
        for (int d = 0; d < 3; ++d) {
            int bit = (s >> d) & 1;
            w *= bit ? fr[d] : (1.0f - fr[d]);
            k += (lo[d] + bit) * mul;
            mul *= KSP;
        }
        bw[s] = w; kx[s] = k;
    }
}

__device__ __forceinline__ float elu1(float v) {
    return v > 0.0f ? v : (expf(v) - 1.0f);
}

// ---------------------------------------------------------------------------
// Layer 1 (Cin = 1): one wave per edge, lane = output channel
// ---------------------------------------------------------------------------
__global__ void conv1_edges(const float* __restrict__ x, const int* __restrict__ ei,
                            const float* __restrict__ ps, const float* __restrict__ W1,
                            float* __restrict__ agg, float* __restrict__ deg) {
    int wid  = (blockIdx.x * blockDim.x + threadIdx.x) >> 5;
    int lane = threadIdx.x & 31;
    if (wid >= E0_) return;
    int src = ei[wid], dst = ei[E0_ + wid];
    float bw[8]; int kx[8];
    edge_basis(ps, wid, bw, kx);
    float m = 0.0f;
#pragma unroll
    for (int s = 0; s < 8; ++s) m += bw[s] * W1[kx[s] * C1_ + lane];
    atomicAdd(&agg[(size_t)dst * C1_ + lane], x[src] * m);
    if (lane == 0) atomicAdd(&deg[dst], 1.0f);
}

// ---------------------------------------------------------------------------
// Binning: histogram by kernel index, padded prefix-sum, scatter records
// record = {src, dst, k, bw_as_int_bits}
// ---------------------------------------------------------------------------
__global__ void count_edges(const float* __restrict__ ps, int E, int* __restrict__ cnt) {
    int e = blockIdx.x * blockDim.x + threadIdx.x;
    if (e >= E) return;
    float bw[8]; int kx[8];
    edge_basis(ps, e, bw, kx);
#pragma unroll
    for (int s = 0; s < 8; ++s) atomicAdd(&cnt[kx[s]], 1);
}

__global__ void scan_offsets(const int* __restrict__ cnt, int* __restrict__ offs) {
    // 125 bins: trivial serial scan, pad each bin to a multiple of 16 rows
    int run = 0;
    for (int k = 0; k < KC; ++k) {
        offs[k] = run;
        run += ((cnt[k] + 15) >> 4) << 4;
    }
}

__global__ void scatter_edges(const int* __restrict__ ei, const float* __restrict__ ps,
                              int E, const int* __restrict__ offs, int* __restrict__ cur,
                              int4* __restrict__ rec, float* __restrict__ deg) {
    int e = blockIdx.x * blockDim.x + threadIdx.x;
    if (e >= E) return;
    int src = ei[e], dst = ei[E + e];
    float bw[8]; int kx[8];
    edge_basis(ps, e, bw, kx);
#pragma unroll
    for (int s = 0; s < 8; ++s) {
        int k   = kx[s];
        int pos = offs[k] + atomicAdd(&cur[k], 1);
        rec[pos] = make_int4(src, dst, k, __float_as_int(bw[s]));
    }
    atomicAdd(&deg[dst], 1.0f);
}

// ---------------------------------------------------------------------------
// Segment GEMM via WMMA: each 16-record tile shares one W_k.
//   A[16 x Cin]  = gathered source-node features (pre-converted f16)
//   B[Cin x 16]  = W_k column tile from the f16 *transposed* weights
// Operand element mapping (ISA 16-bit 16x32 layout):
//   lane L, element i:  row/col = L&15,  K = (i<8?0:16) + (L&16?8:0) + (i&7)
// -> per lane per K-step: two contiguous 16B chunks for A and for B
//    (global_load_b128), no converts in the hot loop.
// C/D layout: VGPR r, lane L: M = r + (L>=16 ? 8 : 0), N = L&15
// Block = 1 wave (EXEC all ones around WMMA); grid = (capTiles, Cout/16).
// ---------------------------------------------------------------------------
__global__ void seg_gemm_wmma(const _Float16* __restrict__ X16,
                              const _Float16* __restrict__ Wt16,
                              const int4* __restrict__ rec, float* __restrict__ agg,
                              int Cin, int Cout) {
    __shared__ int4 r_s[16];
    const int lane = threadIdx.x;
    const int tile = blockIdx.x;
    const int ct   = blockIdx.y;
    if (lane < 16) r_s[lane] = rec[tile * 16 + lane];
    // prefetch next tile's records (global_prefetch_b8); speculative is fine
    if (tile + 1 < (int)gridDim.x)
        __builtin_prefetch(&rec[(tile + 1) * 16], 0, 0);
    __syncthreads();

    const int kk  = r_s[0].z;       // kernel index for this tile (pads: k=0, bw=0)
    const int row = lane & 15;
    const int hi8 = (lane & 16) ? 8 : 0;
    const int src = r_s[row].x;
    const int col = ct * 16 + row;
    const _Float16* Arow = X16  + (size_t)src * Cin + hi8;
    const _Float16* Brow = Wt16 + ((size_t)kk * Cout + col) * Cin + hi8;

    f32x8 acc = {};
    for (int k0 = 0; k0 < Cin; k0 += 32) {
        f16x8 a0 = *(const f16x8*)(Arow + k0);
        f16x8 a1 = *(const f16x8*)(Arow + k0 + 16);
        f16x8 b0 = *(const f16x8*)(Brow + k0);
        f16x8 b1 = *(const f16x8*)(Brow + k0 + 16);
        f16x16 a = __builtin_shufflevector(a0, a1, 0,1,2,3,4,5,6,7,8,9,10,11,12,13,14,15);
        f16x16 b = __builtin_shufflevector(b0, b1, 0,1,2,3,4,5,6,7,8,9,10,11,12,13,14,15);
        acc = __builtin_amdgcn_wmma_f32_16x16x32_f16(false, a, false, b,
                                                     (short)0, acc, false, false);
    }
#pragma unroll
    for (int r = 0; r < 8; ++r) {
        int   m   = r + hi8;
        float bwv = __int_as_float(r_s[m].w);
        int   dst = r_s[m].y;
        atomicAdd(&agg[(size_t)dst * Cout + col], acc[r] * bwv);
    }
}

// ---------------------------------------------------------------------------
// Conv epilogue: mean aggregate + root GEMM + bias + ELU
// ---------------------------------------------------------------------------
__global__ void conv_finalize(const float* __restrict__ agg, const float* __restrict__ deg,
                              const float* __restrict__ xin, const float* __restrict__ root,
                              const float* __restrict__ bias, float* __restrict__ h,
                              int N, int Cin, int Cout) {
    int t = blockIdx.x * blockDim.x + threadIdx.x;
    if (t >= N * Cout) return;
    int n = t / Cout, o = t - n * Cout;
    float d = deg[n]; d = d > 1.0f ? d : 1.0f;
    float v = agg[t] / d;
    for (int c = 0; c < Cin; ++c) v += xin[(size_t)n * Cin + c] * root[(size_t)c * Cout + o];
    v += bias[o];
    h[t] = elu1(v);
}

// ---------------------------------------------------------------------------
// Max pool via order-preserving uint encoding + atomicMax; empty -> 0
// ---------------------------------------------------------------------------
__device__ __forceinline__ unsigned f2ord(float f) {
    unsigned u = __float_as_uint(f);
    return (u & 0x80000000u) ? ~u : (u | 0x80000000u);
}
__global__ void pool_scatter(const float* __restrict__ h, const int* __restrict__ cl,
                             unsigned* __restrict__ poolu, int N, int C) {
    int t = blockIdx.x * blockDim.x + threadIdx.x;
    if (t >= N * C) return;
    int n = t / C, c = t - n * C;
    atomicMax(&poolu[(size_t)cl[n] * C + c], f2ord(h[t]));
}
__global__ void pool_convert(const unsigned* __restrict__ poolu, float* __restrict__ p, int n) {
    int t = blockIdx.x * blockDim.x + threadIdx.x;
    if (t >= n) return;
    unsigned u = poolu[t];
    p[t] = (u == 0u) ? 0.0f
         : ((u & 0x80000000u) ? __uint_as_float(u & 0x7FFFFFFFu) : __uint_as_float(~u));
}

// ---------------------------------------------------------------------------
// FC head prep: pad A to 16 rows (f16), transpose fc1_w to [512][2048] f16
// ---------------------------------------------------------------------------
__global__ void build_fc1_A(const float* __restrict__ p4, _Float16* __restrict__ A16) {
    int t = blockIdx.x * blockDim.x + threadIdx.x;
    if (t >= 16 * 2048) return;
    int row = t >> 11;
    A16[t] = (row < 8) ? (_Float16)p4[t] : (_Float16)0.0f;
}
__global__ void fc1_w_t(const float* __restrict__ W, _Float16* __restrict__ Wt) {
    int t = blockIdx.x * blockDim.x + threadIdx.x;
    if (t >= 2048 * 512) return;
    int K = t / 512, co = t - K * 512;
    Wt[(size_t)co * 2048 + K] = (_Float16)W[t];
}

// ---------------------------------------------------------------------------
// FC1: [8(->16), 2048] x [2048, 512] WMMA GEMM + bias + ELU
// grid = 32 col-tiles, block = 1 wave
// ---------------------------------------------------------------------------
__global__ void fc1_wmma(const _Float16* __restrict__ A16, const _Float16* __restrict__ Wt16,
                         const float* __restrict__ bias, float* __restrict__ out) {
    const int lane = threadIdx.x;
    const int ct   = blockIdx.x;
    const int row  = lane & 15;
    const int hi8  = (lane & 16) ? 8 : 0;
    const int col  = ct * 16 + row;
    const _Float16* Arow = A16  + (size_t)row * 2048 + hi8;
    const _Float16* Brow = Wt16 + (size_t)col * 2048 + hi8;

    f32x8 acc = {};
    for (int k0 = 0; k0 < 2048; k0 += 32) {
        f16x8 a0 = *(const f16x8*)(Arow + k0);
        f16x8 a1 = *(const f16x8*)(Arow + k0 + 16);
        f16x8 b0 = *(const f16x8*)(Brow + k0);
        f16x8 b1 = *(const f16x8*)(Brow + k0 + 16);
        f16x16 a = __builtin_shufflevector(a0, a1, 0,1,2,3,4,5,6,7,8,9,10,11,12,13,14,15);
        f16x16 b = __builtin_shufflevector(b0, b1, 0,1,2,3,4,5,6,7,8,9,10,11,12,13,14,15);
        acc = __builtin_amdgcn_wmma_f32_16x16x32_f16(false, a, false, b,
                                                     (short)0, acc, false, false);
    }
#pragma unroll
    for (int r = 0; r < 8; ++r) {
        int m = r + hi8;
        if (m < 8) out[m * 512 + col] = elu1(acc[r] + bias[col]);
    }
}

// ---------------------------------------------------------------------------
// FC2 + log_softmax (tiny)
// ---------------------------------------------------------------------------
__global__ void fc2_kernel(const float* __restrict__ h, const float* __restrict__ W,
                           const float* __restrict__ b, float* __restrict__ logits) {
    int t = blockIdx.x * blockDim.x + threadIdx.x;
    if (t >= 80) return;
    int r = t / 10, o = t - r * 10;
    float v = b[o];
    for (int j = 0; j < 512; ++j) v += h[r * 512 + j] * W[j * 10 + o];
    logits[r * 10 + o] = v;
}
__global__ void logsoftmax_kernel(const float* __restrict__ logits, float* __restrict__ out) {
    int r = threadIdx.x;
    if (r >= 8) return;
    float m = -1e30f;
    for (int j = 0; j < 10; ++j) m = fmaxf(m, logits[r * 10 + j]);
    float s = 0.0f;
    for (int j = 0; j < 10; ++j) s += expf(logits[r * 10 + j] - m);
    float ls = logf(s);
    for (int j = 0; j < 10; ++j) out[r * 10 + j] = logits[r * 10 + j] - m - ls;
}

// ---------------------------------------------------------------------------
// Workspace layout (bytes)
// ---------------------------------------------------------------------------
#define AL256(x) (((x) + (size_t)255) & ~(size_t)255)
static const size_t REC_CAP_MAX = (size_t)E1_ * 8 + 2000;       // layer 2 largest
static const size_t W16T_MAX    = (size_t)KC * C3_ * C4_;       // layer 4 largest
static const size_t X16_MAX     = (size_t)N1_ * C1_;            // layer 2 input largest

struct WsPlan {
    size_t h0, agg, deg, p1, h1, p2, h2, p3, h3, p4;
    size_t poolu, cnt, offs, cur, rec;
    size_t x16, w16t, a16, fw16t;
    size_t out1, logi, total;
};
static WsPlan plan_ws() {
    WsPlan w; size_t o = 0;
    w.h0    = o; o = AL256(o + (size_t)N0_ * C1_ * 4);
    w.agg   = o; o = AL256(o + (size_t)N0_ * C1_ * 4);  // max agg across layers
    w.deg   = o; o = AL256(o + (size_t)N0_ * 4);
    w.p1    = o; o = AL256(o + (size_t)N1_ * C1_ * 4);
    w.h1    = o; o = AL256(o + (size_t)N1_ * C2_ * 4);
    w.p2    = o; o = AL256(o + (size_t)N2_ * C2_ * 4);
    w.h2    = o; o = AL256(o + (size_t)N2_ * C3_ * 4);
    w.p3    = o; o = AL256(o + (size_t)N3_ * C3_ * 4);
    w.h3    = o; o = AL256(o + (size_t)N3_ * C4_ * 4);
    w.p4    = o; o = AL256(o + (size_t)NV_ * C4_ * 4);
    w.poolu = o; o = AL256(o + (size_t)N1_ * C1_ * 4); // max pool output
    w.cnt   = o; o = AL256(o + (size_t)KC * 4);
    w.offs  = o; o = AL256(o + (size_t)KC * 4);
    w.cur   = o; o = AL256(o + (size_t)KC * 4);
    w.rec   = o; o = AL256(o + REC_CAP_MAX * 16);
    w.x16   = o; o = AL256(o + X16_MAX * 2);
    w.w16t  = o; o = AL256(o + W16T_MAX * 2);
    w.a16   = o; o = AL256(o + (size_t)16 * 2048 * 2);
    w.fw16t = o; o = AL256(o + (size_t)2048 * 512 * 2);
    w.out1  = o; o = AL256(o + (size_t)8 * 512 * 4);
    w.logi  = o; o = AL256(o + (size_t)80 * 4);
    w.total = o;
    return w;
}

static inline int cdiv(int a, int b) { return (a + b - 1) / b; }

static void run_conv_layer(const int* ei, const float* ps, int E,
                           const float* W, const float* root, const float* bias,
                           const float* xin, int N, int Cin, int Cout,
                           float* agg, float* deg, int* cnt, int* offs, int* cur,
                           int4* rec, _Float16* x16, _Float16* w16t,
                           float* h, hipStream_t stream) {
    const int cap = E * 8 + 2000;  // E*8 real records + worst-case pad (125*16)
    // f16 operand prep
    cvt_f16<<<cdiv(N * Cin, 256), 256, 0, stream>>>(xin, x16, N * Cin);
    w_cvt_t<<<cdiv(KC * Cin * Cout, 256), 256, 0, stream>>>(W, w16t, Cin, Cout,
                                                            KC * Cin * Cout);
    // bin edge-corner records by kernel index
    fill_f32<<<cdiv(N * Cout, 256), 256, 0, stream>>>(agg, 0.0f, N * Cout);
    fill_f32<<<cdiv(N, 256), 256, 0, stream>>>(deg, 0.0f, N);
    fill_i32<<<1, 256, 0, stream>>>(cnt, 0, KC);
    fill_i32<<<1, 256, 0, stream>>>(cur, 0, KC);
    fill_rec<<<cdiv(cap, 256), 256, 0, stream>>>(rec, cap);
    count_edges<<<cdiv(E, 256), 256, 0, stream>>>(ps, E, cnt);
    scan_offsets<<<1, 1, 0, stream>>>(cnt, offs);
    scatter_edges<<<cdiv(E, 256), 256, 0, stream>>>(ei, ps, E, offs, cur, rec, deg);
    // dense 16-row tiles x shared W_k -> WMMA
    dim3 g(cap / 16, Cout / 16);
    seg_gemm_wmma<<<g, 32, 0, stream>>>(x16, w16t, rec, agg, Cin, Cout);
    conv_finalize<<<cdiv(N * Cout, 256), 256, 0, stream>>>(agg, deg, xin, root, bias, h,
                                                           N, Cin, Cout);
}

static void run_pool(const float* h, const int* cl, int N, int C, int Nout,
                     unsigned* poolu, float* p, hipStream_t stream) {
    fill_u32<<<cdiv(Nout * C, 256), 256, 0, stream>>>(poolu, 0u, Nout * C);
    pool_scatter<<<cdiv(N * C, 256), 256, 0, stream>>>(h, cl, poolu, N, C);
    pool_convert<<<cdiv(Nout * C, 256), 256, 0, stream>>>(poolu, p, Nout * C);
}

// ---------------------------------------------------------------------------
// Entry point
// ---------------------------------------------------------------------------
extern "C" void kernel_launch(void* const* d_in, const int* in_sizes, int n_in,
                              void* d_out, int out_size, void* d_ws, size_t ws_size,
                              hipStream_t stream) {
    (void)in_sizes; (void)n_in; (void)out_size;
    const float* x    = (const float*)d_in[0];
    const int*   ei1  = (const int*)d_in[1];
    const float* ps1  = (const float*)d_in[2];
    const int*   cl1  = (const int*)d_in[3];
    const int*   ei2  = (const int*)d_in[4];
    const float* ps2  = (const float*)d_in[5];
    const int*   cl2  = (const int*)d_in[6];
    const int*   ei3  = (const int*)d_in[7];
    const float* ps3  = (const float*)d_in[8];
    const int*   cl3  = (const int*)d_in[9];
    const int*   ei4  = (const int*)d_in[10];
    const float* ps4  = (const float*)d_in[11];
    const int*   cl4  = (const int*)d_in[12];
    const float* W1   = (const float*)d_in[13];
    const float* r1   = (const float*)d_in[14];
    const float* b1   = (const float*)d_in[15];
    const float* W2   = (const float*)d_in[16];
    const float* r2   = (const float*)d_in[17];
    const float* b2   = (const float*)d_in[18];
    const float* W3   = (const float*)d_in[19];
    const float* r3   = (const float*)d_in[20];
    const float* b3   = (const float*)d_in[21];
    const float* W4   = (const float*)d_in[22];
    const float* r4   = (const float*)d_in[23];
    const float* b4   = (const float*)d_in[24];
    const float* fw1  = (const float*)d_in[25];
    const float* fb1  = (const float*)d_in[26];
    const float* fw2  = (const float*)d_in[27];
    const float* fb2  = (const float*)d_in[28];

    WsPlan wp = plan_ws();
    if (ws_size < wp.total) return;  // deterministic no-op if scratch too small
    char* ws = (char*)d_ws;
    float*     h0    = (float*)(ws + wp.h0);
    float*     agg   = (float*)(ws + wp.agg);
    float*     deg   = (float*)(ws + wp.deg);
    float*     p1    = (float*)(ws + wp.p1);
    float*     h1    = (float*)(ws + wp.h1);
    float*     p2    = (float*)(ws + wp.p2);
    float*     h2    = (float*)(ws + wp.h2);
    float*     p3    = (float*)(ws + wp.p3);
    float*     h3    = (float*)(ws + wp.h3);
    float*     p4    = (float*)(ws + wp.p4);
    unsigned*  poolu = (unsigned*)(ws + wp.poolu);
    int*       cnt   = (int*)(ws + wp.cnt);
    int*       offs  = (int*)(ws + wp.offs);
    int*       cur   = (int*)(ws + wp.cur);
    int4*      rec   = (int4*)(ws + wp.rec);
    _Float16*  x16   = (_Float16*)(ws + wp.x16);
    _Float16*  w16t  = (_Float16*)(ws + wp.w16t);
    _Float16*  a16   = (_Float16*)(ws + wp.a16);
    _Float16*  fw16t = (_Float16*)(ws + wp.fw16t);
    float*     out1  = (float*)(ws + wp.out1);
    float*     logi  = (float*)(ws + wp.logi);
    float*     out   = (float*)d_out;

    // --- Layer 1 (Cin=1, scalar path) ---
    fill_f32<<<cdiv(N0_ * C1_, 256), 256, 0, stream>>>(agg, 0.0f, N0_ * C1_);
    fill_f32<<<cdiv(N0_, 256), 256, 0, stream>>>(deg, 0.0f, N0_);
    conv1_edges<<<cdiv(E0_ * 32, 256), 256, 0, stream>>>(x, ei1, ps1, W1, agg, deg);
    conv_finalize<<<cdiv(N0_ * C1_, 256), 256, 0, stream>>>(agg, deg, x, r1, b1, h0,
                                                            N0_, 1, C1_);
    run_pool(h0, cl1, N0_, C1_, N1_, poolu, p1, stream);

    // --- Layers 2-4 (binned segment-GEMM via WMMA, f16 operands) ---
    run_conv_layer(ei2, ps2, E1_, W2, r2, b2, p1, N1_, C1_, C2_,
                   agg, deg, cnt, offs, cur, rec, x16, w16t, h1, stream);
    run_pool(h1, cl2, N1_, C2_, N2_, poolu, p2, stream);

    run_conv_layer(ei3, ps3, E2_, W3, r3, b3, p2, N2_, C2_, C3_,
                   agg, deg, cnt, offs, cur, rec, x16, w16t, h2, stream);
    run_pool(h2, cl3, N2_, C3_, N3_, poolu, p3, stream);

    run_conv_layer(ei4, ps4, E3_, W4, r4, b4, p3, N3_, C3_, C4_,
                   agg, deg, cnt, offs, cur, rec, x16, w16t, h3, stream);
    run_pool(h3, cl4, N3_, C4_, NV_, poolu, p4, stream);

    // --- FC head ---
    // p4 is [64, 256] row-major == [8, 2048] row-major after reshape
    build_fc1_A<<<cdiv(16 * 2048, 256), 256, 0, stream>>>(p4, a16);
    fc1_w_t<<<cdiv(2048 * 512, 256), 256, 0, stream>>>(fw1, fw16t);
    fc1_wmma<<<32, 32, 0, stream>>>(a16, fw16t, fb1, out1);
    fc2_kernel<<<1, 128, 0, stream>>>(out1, fw2, fb2, logi);
    logsoftmax_kernel<<<1, 32, 0, stream>>>(logi, out);
}
